// AttentionLayer_90907277787845
// MI455X (gfx1250) — matrix operand order repaired and checked by reading, not compiled, not run on hardware
//
#include <hip/hip_runtime.h>
#include <hip/hip_bf16.h>

// ---------------------------------------------------------------------------
// CDNA5 (gfx1250) attention layer: bf16 WMMA + TDM (tensor_load_to_lds).
// ---------------------------------------------------------------------------

typedef __attribute__((ext_vector_type(16))) __bf16 v16bf;
typedef __attribute__((ext_vector_type(8)))  float  v8f;
typedef __attribute__((ext_vector_type(4)))  unsigned int u32x4;
typedef __attribute__((ext_vector_type(8)))  int    i32x8;
typedef __attribute__((ext_vector_type(4)))  int    i32x4;

union ABFrag { v16bf v; uint4 u[2]; };

#if defined(__has_builtin)
#if __has_builtin(__builtin_amdgcn_tensor_load_to_lds) && \
    __has_builtin(__builtin_amdgcn_s_wait_tensorcnt)
#define HAVE_TDM 1
#endif
#endif
#ifndef HAVE_TDM
#define HAVE_TDM 0
#endif

static __device__ inline unsigned short f2bf(float f) {
    unsigned u = __float_as_uint(f);
    unsigned r = u + 0x7FFFu + ((u >> 16) & 1u);   // round-to-nearest-even
    return (unsigned short)(r >> 16);
}

static __device__ inline v8f wmma_bf16(v16bf a, v16bf b, v8f c) {
    return __builtin_amdgcn_wmma_f32_16x16x32_bf16(
        /*neg_a=*/false, a, /*neg_b=*/false, b,
        /*c_mod=*/(short)0, c, /*reuse_a=*/false, /*reuse_b=*/false);
}

#if HAVE_TDM
// Issue a TDM 2D tile load (bf16 data): tile_d0 elems per row, tile_d1 rows,
// row stride in elements. lds_off = byte offset of destination in LDS.
static __device__ inline void tdm_load_2d(unsigned lds_off, const void* gptr,
                                          unsigned tile_d0, unsigned tile_d1,
                                          unsigned long long row_stride) {
    unsigned long long ga = (unsigned long long)(uintptr_t)gptr;
    u32x4 g0;
    g0[0] = 1u;                                          // count=1 (valid), user mode
    g0[1] = lds_off;                                     // lds_addr
    g0[2] = (unsigned)(ga & 0xFFFFFFFFu);                // global_addr[31:0]
    g0[3] = (unsigned)((ga >> 32) & 0x01FFFFFFu) | (2u << 30);  // addr[56:32], type=2
    // group1 bit assembly (bf16 => data_size code 1)
    unsigned long long a0 = 0x00010000ull                          // data_size=1
        | ((unsigned long long)(tile_d0 & 0xFFFFu) << 48);         // tensor_dim0[15:0]
    unsigned long long a1 = (unsigned long long)((tile_d0 >> 16) & 0xFFFFu)
        | ((unsigned long long)tile_d1 << 16)                      // tensor_dim1
        | ((unsigned long long)(tile_d0 & 0xFFFFu) << 48);         // tile_dim0
    unsigned long long a2 = (unsigned long long)(tile_d1 & 0xFFFFu)  // tile_dim1
        | ((row_stride & 0xFFFFFFFFull) << 32);                    // dim0_stride lo
    unsigned long long a3 = (row_stride >> 32) & 0xFFFFull;        // dim0_stride hi
    i32x8 g1;
    g1[0]=(int)a0; g1[1]=(int)(a0>>32); g1[2]=(int)a1; g1[3]=(int)(a1>>32);
    g1[4]=(int)a2; g1[5]=(int)(a2>>32); g1[6]=(int)a3; g1[7]=(int)(a3>>32);
    i32x4 z4 = {0, 0, 0, 0};
#if __clang_major__ >= 23
    i32x8 z8 = {0, 0, 0, 0, 0, 0, 0, 0};
    __builtin_amdgcn_tensor_load_to_lds(g0, g1, z4, z4, z8, 0);
#else
    __builtin_amdgcn_tensor_load_to_lds(g0, g1, z4, z4, 0);
#endif
}
#endif

// ---------------------------------------------------------------------------
// GEMM: Y[M,N] = X[M,K] * W[N,K]^T.  X is f32 or bf16 (x_bf16 flag),
// W is f32 (converted to bf16 during LDS staging).  Writes f32 and/or bf16.
// Block = 256 threads (8 waves), 128x128 output tile, k-step 32.
// bf16-X tiles are staged by the Tensor Data Mover when available.
// ---------------------------------------------------------------------------
__global__ __launch_bounds__(256) void gemm_xwT_kernel(
    const void* __restrict__ Xv, int x_bf16,
    const float* __restrict__ W,
    float* __restrict__ Yf, unsigned short* __restrict__ Ybf,
    int M, int N, int K)
{
    __shared__ __align__(16) unsigned short Xs[128 * 32];
    __shared__ __align__(16) unsigned short Ws[128 * 32];

    const int tid  = threadIdx.x;
    const int lane = tid & 31;
    const int w    = tid >> 5;
    const int l15  = lane & 15;
    const int lhi  = lane >> 4;
    const int wm   = (w & 3) * 32;   // 4 wave-rows of 32
    const int wn   = (w >> 2) * 64;  // 2 wave-cols of 64
    const int m0   = blockIdx.y * 128;
    const int n0   = blockIdx.x * 128;

    v8f acc[2][4] = {};

    for (int k0 = 0; k0 < K; k0 += 32) {
        __syncthreads();
        if (x_bf16) {
            const unsigned short* Xb = (const unsigned short*)Xv;
#if HAVE_TDM
            if (w == 0)
                tdm_load_2d((unsigned)(uintptr_t)&Xs[0],
                            &Xb[(size_t)m0 * K + k0], 32, 128,
                            (unsigned long long)K);
#else
            #pragma unroll
            for (int i = 0; i < 2; ++i) {
                int e = tid * 8 + i * 2048;          // 4096 bf16 elems total
                int row = e >> 5, col = e & 31;
                *(uint4*)&Xs[row * 32 + col] =
                    *(const uint4*)&Xb[(size_t)(m0 + row) * K + k0 + col];
            }
#endif
        } else {
            const float* Xf = (const float*)Xv;
            #pragma unroll
            for (int i = 0; i < 4; ++i) {
                int e = tid * 4 + i * 1024;
                int row = e >> 5, col = e & 31;
                float4 f = *(const float4*)&Xf[(size_t)(m0 + row) * K + k0 + col];
                uint2 p;
                p.x = (unsigned)f2bf(f.x) | ((unsigned)f2bf(f.y) << 16);
                p.y = (unsigned)f2bf(f.z) | ((unsigned)f2bf(f.w) << 16);
                *(uint2*)&Xs[row * 32 + col] = p;
            }
        }
        #pragma unroll
        for (int i = 0; i < 4; ++i) {
            int e = tid * 4 + i * 1024;
            int row = e >> 5, col = e & 31;      // row indexes N dimension of W
            float4 f = *(const float4*)&W[(size_t)(n0 + row) * K + k0 + col];
            uint2 p;
            p.x = (unsigned)f2bf(f.x) | ((unsigned)f2bf(f.y) << 16);
            p.y = (unsigned)f2bf(f.z) | ((unsigned)f2bf(f.w) << 16);
            *(uint2*)&Ws[row * 32 + col] = p;
        }
#if HAVE_TDM
        if (x_bf16) __builtin_amdgcn_s_wait_tensorcnt(0);
#endif
        __syncthreads();

        #pragma unroll
        for (int mt = 0; mt < 2; ++mt) {
            ABFrag a;
            int arow = wm + mt * 16 + l15;
            a.u[0] = *(const uint4*)&Xs[arow * 32 + lhi * 8];
            a.u[1] = *(const uint4*)&Xs[arow * 32 + lhi * 8 + 16];
            #pragma unroll
            for (int nt = 0; nt < 4; ++nt) {
                ABFrag bb;
                int brow = wn + nt * 16 + l15;
                bb.u[0] = *(const uint4*)&Ws[brow * 32 + lhi * 8];
                bb.u[1] = *(const uint4*)&Ws[brow * 32 + lhi * 8 + 16];
                acc[mt][nt] = wmma_bf16(a.v, bb.v, acc[mt][nt]);
            }
        }
    }

    #pragma unroll
    for (int mt = 0; mt < 2; ++mt) {
        #pragma unroll
        for (int nt = 0; nt < 4; ++nt) {
            int col  = n0 + wn + nt * 16 + l15;
            int rowb = m0 + wm + mt * 16 + lhi * 8;
            #pragma unroll
            for (int r = 0; r < 8; ++r) {
                float val = acc[mt][nt][r];
                size_t idx = (size_t)(rowb + r) * N + col;
                if (Yf)  Yf[idx]  = val;
                if (Ybf) Ybf[idx] = f2bf(val);
            }
        }
    }
}

// ---------------------------------------------------------------------------
// Flash attention: one block per (q-tile of 128, head, batch).
// 8 waves x 16 query rows.  K-tile = 64 keys.  Dh = 128.
// Kt staged by TDM (row-major [key][dh]); Vt transposed manually [dh][key]
// so both WMMA B-fragment reads are two 16-byte loads per lane within one row.
// ---------------------------------------------------------------------------
__global__ __launch_bounds__(256) void attn_kernel(
    const unsigned short* __restrict__ Qb,
    const unsigned short* __restrict__ Kb,
    const unsigned short* __restrict__ Vb,
    const unsigned char*  __restrict__ mask,
    unsigned short* __restrict__ Ctx,
    int S, int D)
{
    __shared__ __align__(16) unsigned short Kt[64 * 128];   // [key][dh]
    __shared__ __align__(16) unsigned short Vt[128 * 64];   // [dh][key]
    __shared__ __align__(16) unsigned short Ps[8 * 16 * 64];

    const int tid  = threadIdx.x;
    const int lane = tid & 31;
    const int w    = tid >> 5;
    const int l15  = lane & 15;
    const int lhi  = lane >> 4;
    const int q0   = blockIdx.x * 128;
    const int h    = blockIdx.y;
    const int b    = blockIdx.z;
    const int qrow0 = q0 + w * 16;
    unsigned short* Pw = &Ps[w * 16 * 64];
    const float scale = 0.08838834764831845f;   // 1/sqrt(128)

    // Q fragments (A-layout), kept in registers for the whole kernel.
    ABFrag qf[4];
    {
        size_t base = ((size_t)b * S + qrow0 + l15) * (size_t)D + h * 128 + lhi * 8;
        #pragma unroll
        for (int ks = 0; ks < 4; ++ks) {
            qf[ks].u[0] = *(const uint4*)&Qb[base + ks * 32];
            qf[ks].u[1] = *(const uint4*)&Qb[base + ks * 32 + 16];
        }
    }

    v8f cacc[8] = {};
    float row_m[8], row_l[8];
    #pragma unroll
    for (int r = 0; r < 8; ++r) { row_m[r] = -1e30f; row_l[r] = 0.0f; }

    for (int j = 0; j < S; j += 64) {
        __syncthreads();
        // K tile: TDM DMA (64 rows x 128 bf16, row stride D) into Kt.
#if HAVE_TDM
        if (w == 0)
            tdm_load_2d((unsigned)(uintptr_t)&Kt[0],
                        &Kb[((size_t)b * S + j) * (size_t)D + h * 128],
                        128, 64, (unsigned long long)D);
#endif
        // V tile: manual transpose staging (TDM cannot transpose).
        #pragma unroll
        for (int i = 0; i < 4; ++i) {
            int e   = tid * 8 + i * 2048;            // 8192 bf16 elems
            int key = e >> 7, dh = e & 127;
            size_t src = ((size_t)b * S + j + key) * (size_t)D + h * 128 + dh;
#if !HAVE_TDM
            *(uint4*)&Kt[key * 128 + dh] = *(const uint4*)&Kb[src];
#endif
            union { uint4 u; unsigned short s[8]; } vv;
            vv.u = *(const uint4*)&Vb[src];
            #pragma unroll
            for (int ii = 0; ii < 8; ++ii)
                Vt[(dh + ii) * 64 + key] = vv.s[ii];
        }
#if HAVE_TDM
        __builtin_amdgcn_s_wait_tensorcnt(0);
#endif
        __syncthreads();

        // scores = Q * K^T  (16x64 per wave, f32 accum)
        v8f sacc[4] = {};
        #pragma unroll
        for (int ks = 0; ks < 4; ++ks) {
            #pragma unroll
            for (int nt = 0; nt < 4; ++nt) {
                ABFrag bb;
                int kc = nt * 16 + l15;
                bb.u[0] = *(const uint4*)&Kt[kc * 128 + ks * 32 + lhi * 8];
                bb.u[1] = *(const uint4*)&Kt[kc * 128 + ks * 32 + lhi * 8 + 16];
                sacc[nt] = wmma_bf16(qf[ks].v, bb.v, sacc[nt]);
            }
        }

        // scale + additive mask
        int rq0 = qrow0 + lhi * 8;
        #pragma unroll
        for (int nt = 0; nt < 4; ++nt) {
            int col = j + nt * 16 + l15;
            #pragma unroll
            for (int r = 0; r < 8; ++r) {
                float s = sacc[nt][r] * scale;
                s += (float)mask[((size_t)b * S + rq0 + r) * (size_t)S + col] * (-10000.0f);
                sacc[nt][r] = s;
            }
        }

        // online softmax per query row (rows live in 16-lane half-wave groups)
        #pragma unroll
        for (int r = 0; r < 8; ++r) {
            float mx = fmaxf(fmaxf(sacc[0][r], sacc[1][r]),
                             fmaxf(sacc[2][r], sacc[3][r]));
            #pragma unroll
            for (int d = 1; d < 16; d <<= 1)
                mx = fmaxf(mx, __shfl_xor(mx, d, 32));
            float mn   = fmaxf(row_m[r], mx);
            float corr = __expf(row_m[r] - mn);
            row_m[r] = mn;
            float sum = 0.0f;
            #pragma unroll
            for (int nt = 0; nt < 4; ++nt) {
                float p = __expf(sacc[nt][r] - mn);
                sacc[nt][r] = p;
                sum += p;
            }
            #pragma unroll
            for (int d = 1; d < 16; d <<= 1)
                sum += __shfl_xor(sum, d, 32);
            row_l[r] = row_l[r] * corr + sum;
            #pragma unroll
            for (int dt = 0; dt < 8; ++dt)
                cacc[dt][r] *= corr;
        }

        // P -> per-wave LDS (C-layout store, A-layout reload)
        #pragma unroll
        for (int nt = 0; nt < 4; ++nt)
            #pragma unroll
            for (int r = 0; r < 8; ++r)
                Pw[(r + lhi * 8) * 64 + nt * 16 + l15] = f2bf(sacc[nt][r]);
        __syncthreads();

        // ctx += P * V
        #pragma unroll
        for (int ks2 = 0; ks2 < 2; ++ks2) {
            ABFrag a;
            a.u[0] = *(const uint4*)&Pw[l15 * 64 + ks2 * 32 + lhi * 8];
            a.u[1] = *(const uint4*)&Pw[l15 * 64 + ks2 * 32 + lhi * 8 + 16];
            #pragma unroll
            for (int dt = 0; dt < 8; ++dt) {
                ABFrag bb;
                int dh = dt * 16 + l15;
                bb.u[0] = *(const uint4*)&Vt[dh * 64 + ks2 * 32 + lhi * 8];
                bb.u[1] = *(const uint4*)&Vt[dh * 64 + ks2 * 32 + lhi * 8 + 16];
                cacc[dt] = wmma_bf16(a.v, bb.v, cacc[dt]);
            }
        }
    }

    // normalize and write ctx (bf16) in [b][s][h*128+dh] layout (merged heads)
    #pragma unroll
    for (int dt = 0; dt < 8; ++dt) {
        int col = h * 128 + dt * 16 + l15;
        #pragma unroll
        for (int r = 0; r < 8; ++r) {
            int row = qrow0 + r + lhi * 8;
            float vv = cacc[dt][r] / row_l[r];
            Ctx[((size_t)b * S + row) * (size_t)D + col] = f2bf(vv);
        }
    }
}

// ---------------------------------------------------------------------------
// Launch: inputs (q,k,v,mask,Wq,Wk,Wv,Wo); d_out = [out | kp | vp] (f32).
// Workspace: qb | kb | vb | ctx  (bf16, B*S*D each = 67 MB total).
// ---------------------------------------------------------------------------
extern "C" void kernel_launch(void* const* d_in, const int* in_sizes, int n_in,
                              void* d_out, int out_size, void* d_ws, size_t ws_size,
                              hipStream_t stream) {
    (void)in_sizes; (void)n_in; (void)out_size; (void)ws_size;
    const float* q  = (const float*)d_in[0];
    const float* k  = (const float*)d_in[1];
    const float* v  = (const float*)d_in[2];
    const unsigned char* mask = (const unsigned char*)d_in[3];
    const float* Wq = (const float*)d_in[4];
    const float* Wk = (const float*)d_in[5];
    const float* Wv = (const float*)d_in[6];
    const float* Wo = (const float*)d_in[7];
    float* out = (float*)d_out;

    const int B = 2, S = 2048, D = 2048, H = 16;
    const size_t BSD = (size_t)B * S * D;

    unsigned short* qb   = (unsigned short*)d_ws;
    unsigned short* kb   = qb + BSD;
    unsigned short* vb   = kb + BSD;
    unsigned short* ctxb = vb + BSD;

    dim3 gg(D / 128, (B * S) / 128);   // (16, 32)

    // Projections: qp (bf16 only), kp/vp (f32 to d_out + bf16 to ws)
    gemm_xwT_kernel<<<gg, 256, 0, stream>>>(q, 0, Wq, nullptr,      qb, B * S, D, D);
    gemm_xwT_kernel<<<gg, 256, 0, stream>>>(k, 0, Wk, out + BSD,    kb, B * S, D, D);
    gemm_xwT_kernel<<<gg, 256, 0, stream>>>(v, 0, Wv, out + 2*BSD,  vb, B * S, D, D);

    // Flash attention
    attn_kernel<<<dim3(S / 128, H, B), 256, 0, stream>>>(qb, kb, vb, mask, ctxb, S, D);

    // Output projection: out = ctx @ Wo^T
    gemm_xwT_kernel<<<gg, 256, 0, stream>>>(ctxb, 1, Wo, out, nullptr, B * S, D, D);
}
